// EfficientDet_77893526880317
// MI455X (gfx1250) — matrix operand order repaired and checked by reading, not compiled, not run on hardware
//
#include <hip/hip_runtime.h>

typedef __attribute__((ext_vector_type(16))) _Float16 v16h;
typedef __attribute__((ext_vector_type(8)))  float    v8f;

#define NA   49104
#define NSEG 768
#define TOPK 1000

// ---------------------------------------------------------------------------
// Pack an OIHW fp32 weight tensor into the CDNA5 WMMA A-fragment layout
// (16x32 f16 per (mtile,ktile), lane l: M = l&15,
//  K = (e>>3)*16 + (l>>4)*8 + (e&7)), contiguous per fragment.
// ---------------------------------------------------------------------------
__global__ void pack_w_kernel(const float* __restrict__ w, _Float16* __restrict__ ap,
                              int O, int Cin, int KH, int KW, int mtiles, int ktiles)
{
    int idx = blockIdx.x * blockDim.x + threadIdx.x;
    int total = mtiles * ktiles * 512;
    if (idx >= total) return;
    int e    = idx & 15;
    int lane = (idx >> 4) & 31;
    int rest = idx >> 9;
    int kt = rest % ktiles;
    int mt = rest / ktiles;
    int M = mt * 16 + (lane & 15);
    int K = kt * 32 + (e >> 3) * 16 + (lane >> 4) * 8 + (e & 7);
    int taps = KH * KW;
    int Kdim = Cin * taps;
    float v = 0.f;
    if (M < O && K < Kdim) {
        int ci = K / taps;
        int t  = K - ci * taps;
        v = w[(size_t)(M * Cin + ci) * taps + t];
    }
    ap[idx] = (_Float16)v;
}

// ---------------------------------------------------------------------------
// Implicit-GEMM conv, M-blocked 4x: one wave computes a 64(cout) x 16(pixel)
// tile = 4 accumulators fed by ONE shared B fragment per K-step, so the
// expensive per-lane patch gather is amortized over 4 v_wmma_f32_16x16x32_f16.
// Template params make taps and epilogue compile-time; Wout is a power of two
// (shift/mask addressing, no runtime division).
//   EPI: 0 = +bias, 1 = +bias+relu, 2 = BN (y*scale + shift)
// ---------------------------------------------------------------------------
template <int KH, int KW, int RELU_IN, int EPI>
__global__ __launch_bounds__(32)
void conv_wmma_kernel(const float* __restrict__ x, const _Float16* __restrict__ ap,
                      const float* __restrict__ p0, const float* __restrict__ p1,
                      float* __restrict__ out,
                      int Cin, int Cout, int Hin, int Win,
                      int wshift, int npix, int stride, int pad,
                      int ktiles, int mtiles)
{
    constexpr int taps = KH * KW;
    const int lane = threadIdx.x;
    const int nt  = blockIdx.x;
    const int mt0 = blockIdx.y * 4;
    const int ncol = nt * 16 + (lane & 15);      // output pixel (B/D column N)
    const int half = lane >> 4;
    const int wmask = (1 << wshift) - 1;
    const int oy = ncol >> wshift, ox = ncol & wmask;
    const int iy0 = oy * stride - pad, ix0 = ox * stride - pad;
    const size_t plane = (size_t)Hin * Win;

    v8f acc0 = {}, acc1 = {}, acc2 = {}, acc3 = {};
    for (int kt = 0; kt < ktiles; ++kt) {
        // --- B fragment: lane l holds column N = l&15, K = (e>>3)*16 + (l>>4)*8 + (e&7) ---
        v16h bfrag;
        const int kbase = kt * 32 + half * 8;
#pragma unroll
        for (int e = 0; e < 16; ++e) {
            int k  = kbase + (e >> 3) * 16 + (e & 7);
            int ci = k / taps;                    // constexpr divisor
            int t  = k - ci * taps;
            int ky = t / KW;
            int kx = t - ky * KW;
            int iy = iy0 + ky, ix = ix0 + kx;
            float v = 0.f;
            if (ci < Cin && iy >= 0 && iy < Hin && ix >= 0 && ix < Win) {
                v = x[ci * plane + (size_t)iy * Win + ix];
                if (RELU_IN) v = fmaxf(v, 0.f);
            }
            bfrag[e] = (_Float16)v;
        }
        // prefetch next K-step's A block (global_prefetch_b8 path)
        if (kt + 1 < ktiles)
            __builtin_prefetch(ap + ((size_t)(mt0 * ktiles + kt + 1) * 32) * 16, 0, 0);
        // --- 4 A fragments (clamped in-bounds; dead results masked at store) ---
        int m0 = mt0 < mtiles - 1 ? mt0 : mtiles - 1;
        int m1 = mt0 + 1 < mtiles - 1 ? mt0 + 1 : mtiles - 1;
        int m2 = mt0 + 2 < mtiles - 1 ? mt0 + 2 : mtiles - 1;
        int m3 = mt0 + 3 < mtiles - 1 ? mt0 + 3 : mtiles - 1;
        v16h a0 = *(const v16h*)(ap + ((size_t)(m0 * ktiles + kt) * 32 + lane) * 16);
        v16h a1 = *(const v16h*)(ap + ((size_t)(m1 * ktiles + kt) * 32 + lane) * 16);
        v16h a2 = *(const v16h*)(ap + ((size_t)(m2 * ktiles + kt) * 32 + lane) * 16);
        v16h a3 = *(const v16h*)(ap + ((size_t)(m3 * ktiles + kt) * 32 + lane) * 16);
        acc0 = __builtin_amdgcn_wmma_f32_16x16x32_f16(false, a0, false, bfrag, (short)0, acc0, false, false);
        acc1 = __builtin_amdgcn_wmma_f32_16x16x32_f16(false, a1, false, bfrag, (short)0, acc1, false, false);
        acc2 = __builtin_amdgcn_wmma_f32_16x16x32_f16(false, a2, false, bfrag, (short)0, acc2, false, false);
        acc3 = __builtin_amdgcn_wmma_f32_16x16x32_f16(false, a3, false, bfrag, (short)0, acc3, false, false);
    }
    // D layout: VGPR r holds M = (l>>4)*8 + r, N = l&15
    v8f accs[4] = {acc0, acc1, acc2, acc3};
#pragma unroll
    for (int j = 0; j < 4; ++j) {
#pragma unroll
        for (int r = 0; r < 8; ++r) {
            int co = (mt0 + j) * 16 + half * 8 + r;   // unclamped -> guards dupes
            if (co < Cout) {
                float y = accs[j][r];
                if (EPI == 2)      y = y * p0[co] + p1[co];
                else { y += p0[co]; if (EPI == 1) y = fmaxf(y, 0.f); }
                out[(size_t)co * npix + ncol] = y;
            }
        }
    }
}

// ---------------------------------------------------------------------------
// BN folding: scale = g*rsqrt(v+eps); shift = (bias - mean)*scale + beta
// ---------------------------------------------------------------------------
__global__ void bnprep_kernel(const float* __restrict__ b, const float* __restrict__ g,
                              const float* __restrict__ bt, const float* __restrict__ m,
                              const float* __restrict__ v,
                              float* __restrict__ scale, float* __restrict__ shift, int n)
{
    int i = blockIdx.x * blockDim.x + threadIdx.x;
    if (i >= n) return;
    float sc = g[i] * rsqrtf(v[i] + 4e-5f);
    scale[i] = sc;
    shift[i] = (b[i] - m[i]) * sc + bt[i];
}

// out = u0*a + u1*up2(bs);  fast-normalized weights from w[0..1]
__global__ void fuse2_kernel(const float* __restrict__ a, const float* __restrict__ bs,
                             float* __restrict__ out, const float* __restrict__ w,
                             int H, int W)
{
    int idx = blockIdx.x * blockDim.x + threadIdx.x;
    int total = 64 * H * W;
    if (idx >= total) return;
    float u0 = fmaxf(w[0], 0.f), u1 = fmaxf(w[1], 0.f);
    float s = u0 + u1 + 1e-4f; u0 /= s; u1 /= s;
    int x = idx % W; int y = (idx / W) % H; int c = idx / (W * H);
    int Hh = H >> 1, Wh = W >> 1;
    out[idx] = u0 * a[idx] + u1 * bs[(c * Hh + (y >> 1)) * Wh + (x >> 1)];
}

// out = u0*a + u1*b + u2*down2(cl);  cl is at 2H x 2W
__global__ void fuse3_kernel(const float* __restrict__ a, const float* __restrict__ b,
                             const float* __restrict__ cl, float* __restrict__ out,
                             const float* __restrict__ w, int H, int W)
{
    int idx = blockIdx.x * blockDim.x + threadIdx.x;
    int total = 64 * H * W;
    if (idx >= total) return;
    float u0 = fmaxf(w[0], 0.f), u1 = fmaxf(w[1], 0.f), u2 = fmaxf(w[2], 0.f);
    float s = u0 + u1 + u2 + 1e-4f; u0 /= s; u1 /= s; u2 /= s;
    int x = idx % W; int y = (idx / W) % H; int c = idx / (W * H);
    out[idx] = u0 * a[idx] + u1 * b[idx] + u2 * cl[(c * 2 * H + 2 * y) * (2 * W) + 2 * x];
}

// ---------------------------------------------------------------------------
// Per-anchor decode: anchors synthesized analytically; max/argmax over classes;
// score = sigmoid(max logit) (sigmoid is monotone, so commutes with max/argmax)
// ---------------------------------------------------------------------------
__global__ void decode_kernel(const float* __restrict__ r0, const float* __restrict__ r1,
                              const float* __restrict__ r2, const float* __restrict__ r3,
                              const float* __restrict__ r4,
                              const float* __restrict__ q0, const float* __restrict__ q1,
                              const float* __restrict__ q2, const float* __restrict__ q3,
                              const float* __restrict__ q4,
                              float* __restrict__ boxes, float* __restrict__ scores,
                              int* __restrict__ labels)
{
    int i = blockIdx.x * blockDim.x + threadIdx.x;
    if (i >= NA) return;
    int lvl, base;
    if      (i < 36864) { lvl = 0; base = 0;     }
    else if (i < 46080) { lvl = 1; base = 36864; }
    else if (i < 48384) { lvl = 2; base = 46080; }
    else if (i < 48960) { lvl = 3; base = 48384; }
    else                { lvl = 4; base = 48960; }
    const float* rp; const float* cp;
    switch (lvl) {
        case 0:  rp = r0; cp = q0; break;
        case 1:  rp = r1; cp = q1; break;
        case 2:  rp = r2; cp = q2; break;
        case 3:  rp = r3; cp = q3; break;
        default: rp = r4; cp = q4; break;
    }
    int fs = 64 >> lvl;
    int li = i - base;
    int cell = li / 9, a = li - cell * 9;
    int y = cell / fs, xx = cell - y * fs;
    int npix = fs * fs;

    float rg0 = rp[(a * 4 + 0) * npix + cell];
    float rg1 = rp[(a * 4 + 1) * npix + cell];
    float rg2 = rp[(a * 4 + 2) * npix + cell];
    float rg3 = rp[(a * 4 + 3) * npix + cell];

    float best = -3.4e38f; int bl = 0;
    for (int c = 0; c < 80; ++c) {
        float v = cp[(a * 80 + c) * npix + cell];
        if (v > best) { best = v; bl = c; }
    }
    float score = 1.f / (1.f + expf(-best));

    const float sc3[3] = {1.f, 1.2599211f, 1.5874011f};  // 2^(k/3)
    const float rt3[3] = {0.5f, 1.f, 2.f};
    float stride = (float)(8  << lvl);
    float bsz    = (float)(32 << lvl);
    float ss = sc3[a % 3], rr = rt3[a / 3];
    float w0 = bsz * ss;
    float wa = w0 * rsqrtf(rr);     // sqrt(area / r)
    float ha = wa * rr;
    float cxa = (xx + 0.5f) * stride, cya = (y + 0.5f) * stride;
    float cx = cxa + rg0 * 0.1f * wa;
    float cy = cya + rg1 * 0.1f * ha;
    float ww = expf(rg2 * 0.2f) * wa;
    float hh = expf(rg3 * 0.2f) * ha;
    float x1 = fmaxf(cx - 0.5f * ww, 0.f);
    float y1 = fmaxf(cy - 0.5f * hh, 0.f);
    float x2 = fminf(cx + 0.5f * ww, 512.f);
    float y2 = fminf(cy + 0.5f * hh, 512.f);
    boxes[i * 4 + 0] = x1; boxes[i * 4 + 1] = y1;
    boxes[i * 4 + 2] = x2; boxes[i * 4 + 3] = y2;
    scores[i] = score; labels[i] = bl;
}

// ---------------------------------------------------------------------------
// Top-k (k=1000) via hierarchical segment-max: 768 segments of 64;
// each round scans 768 segment maxima instead of 49104 values.
// Tie-break: lowest global index (matches stable jax top_k).
// ---------------------------------------------------------------------------
__global__ __launch_bounds__(256)
void topk_kernel(const float* __restrict__ scores, float* __restrict__ work,
                 float* __restrict__ top_s, int* __restrict__ top_i)
{
    __shared__ float segv[NSEG]; __shared__ int segi[NSEG];
    __shared__ float rv[256];    __shared__ int ri[256];
    int tid = threadIdx.x;
    for (int i = tid; i < NA; i += 256)
        work[i] = scores[i] > 0.05f ? scores[i] : -1.0f;
    __syncthreads();
    for (int s = tid; s < NSEG; s += 256) {
        float bv = -1e30f; int bi = -1;
        int b0 = s * 64, b1 = min(b0 + 64, NA);
        for (int i = b0; i < b1; ++i) { float v = work[i]; if (v > bv) { bv = v; bi = i; } }
        segv[s] = bv; segi[s] = bi;
    }
    __syncthreads();
    for (int t = 0; t < TOPK; ++t) {
        float bv = -1e30f; int bi = 0x7fffffff;
        for (int s = tid; s < NSEG; s += 256) {
            float v = segv[s]; int i = segi[s];
            if (v > bv || (v == bv && i < bi)) { bv = v; bi = i; }
        }
        rv[tid] = bv; ri[tid] = bi; __syncthreads();
        for (int offp = 128; offp > 0; offp >>= 1) {
            if (tid < offp) {
                if (rv[tid + offp] > rv[tid] ||
                    (rv[tid + offp] == rv[tid] && ri[tid + offp] < ri[tid])) {
                    rv[tid] = rv[tid + offp]; ri[tid] = ri[tid + offp];
                }
            }
            __syncthreads();
        }
        int win = ri[0]; float wv = rv[0];
        if (tid == 0) {
            top_s[t] = wv; top_i[t] = win;
            work[win] = -1e30f;
            int s = win / 64; float nb = -1e30f; int nbi = -1;
            int b0 = s * 64, b1 = min(b0 + 64, NA);
            for (int i = b0; i < b1; ++i) { float v = work[i]; if (v > nb) { nb = v; nbi = i; } }
            segv[s] = nb; segi[s] = nbi;
        }
        __syncthreads();
    }
}

// ---------------------------------------------------------------------------
// NMS: all 1000 candidates resident in LDS, sequential-i suppression
// (matches the reference fori_loop), then write final outputs.
// d_out layout: [1000 scores][1000 labels(as float)][1000*4 boxes]
// ---------------------------------------------------------------------------
__global__ __launch_bounds__(1024)
void nms_kernel(const float* __restrict__ boxes, const int* __restrict__ labels,
                const float* __restrict__ top_s, const int* __restrict__ top_i,
                float* __restrict__ out)
{
    __shared__ float bx1[TOPK], by1[TOPK], bx2[TOPK], by2[TOPK], ar[TOPK], sc[TOPK];
    __shared__ int   kp[TOPK], lb[TOPK];
    int tid = threadIdx.x;
    if (tid < TOPK) {
        int gi = top_i[tid];
        float s = top_s[tid];
        float a = boxes[gi * 4 + 0], b = boxes[gi * 4 + 1];
        float c = boxes[gi * 4 + 2], d = boxes[gi * 4 + 3];
        bx1[tid] = a; by1[tid] = b; bx2[tid] = c; by2[tid] = d;
        ar[tid] = fmaxf(c - a, 0.f) * fmaxf(d - b, 0.f);
        sc[tid] = s;
        kp[tid] = (s > 0.05f) ? 1 : 0;
        lb[tid] = labels[gi];
    }
    __syncthreads();
    for (int i = 0; i < TOPK; ++i) {
        if (tid < TOPK && tid > i && kp[tid] && kp[i]) {
            float iw = fmaxf(fminf(bx2[tid], bx2[i]) - fmaxf(bx1[tid], bx1[i]), 0.f);
            float ih = fmaxf(fminf(by2[tid], by2[i]) - fmaxf(by1[tid], by1[i]), 0.f);
            float inter = iw * ih;
            float iou = inter / (ar[tid] + ar[i] - inter + 1e-8f);
            if (iou > 0.5f) kp[tid] = 0;
        }
        __syncthreads();
    }
    if (tid < TOPK) {
        int k = kp[tid];
        out[tid]          = k ? sc[tid] : 0.f;
        out[TOPK + tid]   = k ? (float)lb[tid] : -1.f;
        out[2 * TOPK + tid * 4 + 0] = k ? bx1[tid] : 0.f;
        out[2 * TOPK + tid * 4 + 1] = k ? by1[tid] : 0.f;
        out[2 * TOPK + tid * 4 + 2] = k ? bx2[tid] : 0.f;
        out[2 * TOPK + tid * 4 + 3] = k ? by2[tid] : 0.f;
    }
}

// ---------------------------------------------------------------------------
// Host orchestration
// ---------------------------------------------------------------------------
extern "C" void kernel_launch(void* const* d_in, const int* in_sizes, int n_in,
                              void* d_out, int out_size, void* d_ws, size_t ws_size,
                              hipStream_t stream)
{
    (void)in_sizes; (void)n_in; (void)out_size; (void)ws_size;

    const float* c3        = (const float*)d_in[0];   // (16,40,64,64) -> use image 0
    const float* c4        = (const float*)d_in[1];
    const float* c5        = (const float*)d_in[2];
    const float* in_w3     = (const float*)d_in[3];
    const float* in_b3     = (const float*)d_in[4];
    const float* in_w4     = (const float*)d_in[5];
    const float* in_b4     = (const float*)d_in[6];
    const float* in_w5     = (const float*)d_in[7];
    const float* in_b5     = (const float*)d_in[8];
    const float* in_w6     = (const float*)d_in[9];
    const float* in_b6     = (const float*)d_in[10];
    const float* in_w7     = (const float*)d_in[11];
    const float* in_b7     = (const float*)d_in[12];
    const float* td_w      = (const float*)d_in[13];
    const float* td_b      = (const float*)d_in[14];
    const float* td_g      = (const float*)d_in[15];
    const float* td_beta   = (const float*)d_in[16];
    const float* td_m      = (const float*)d_in[17];
    const float* td_v      = (const float*)d_in[18];
    const float* oc_w      = (const float*)d_in[19];
    const float* oc_b      = (const float*)d_in[20];
    const float* oc_g      = (const float*)d_in[21];
    const float* oc_beta   = (const float*)d_in[22];
    const float* oc_m      = (const float*)d_in[23];
    const float* oc_v      = (const float*)d_in[24];
    const float* fpn_w1    = (const float*)d_in[25];  // (2,2)
    const float* fpn_w2    = (const float*)d_in[26];  // (2,3)
    const float* cls_w     = (const float*)d_in[27];  // (3,64,64,3,3)
    const float* cls_b     = (const float*)d_in[28];
    const float* cls_ow    = (const float*)d_in[29];  // (720,64,3,3)
    const float* cls_ob    = (const float*)d_in[30];
    const float* reg_w     = (const float*)d_in[31];
    const float* reg_b     = (const float*)d_in[32];
    const float* reg_ow    = (const float*)d_in[33];  // (36,64,3,3)
    const float* reg_ob    = (const float*)d_in[34];

    char* base = (char*)d_ws;
    size_t off = 0;
    auto alloc = [&](size_t bytes) -> void* {
        void* p = base + off;
        off += (bytes + 255) & ~(size_t)255;
        return p;
    };

    struct PW { const _Float16* ap; int mt, kt; };
    auto pack = [&](const float* w, int O, int Cin, int KH, int KW) -> PW {
        PW r;
        r.mt = (O + 15) / 16;
        r.kt = (Cin * KH * KW + 31) / 32;
        size_t n = (size_t)r.mt * r.kt * 512;
        _Float16* p = (_Float16*)alloc(n * sizeof(_Float16));
        pack_w_kernel<<<dim3((unsigned)((n + 255) / 256)), dim3(256), 0, stream>>>(
            w, p, O, Cin, KH, KW, r.mt, r.kt);
        r.ap = p;
        return r;
    };
    auto conv = [&](const float* x, PW pw, const float* p0, const float* p1, float* out,
                    int Cin, int Cout, int Hin, int Win, int Hout, int Wout,
                    int KH, int KW, int stride, int pad, int relu_in, int epi) {
        int wshift = (Wout == 64) ? 6 : (Wout == 32) ? 5 : (Wout == 16) ? 4
                   : (Wout == 8)  ? 3 : 2;
        int npix = Hout * Wout;
        dim3 grid((unsigned)((npix + 15) / 16), (unsigned)((pw.mt + 3) / 4));
        if (KH == 1) {
            if (epi == 2)
                conv_wmma_kernel<1, 1, 0, 2><<<grid, dim3(32), 0, stream>>>(
                    x, pw.ap, p0, p1, out, Cin, Cout, Hin, Win, wshift, npix, stride, pad, pw.kt, pw.mt);
            else
                conv_wmma_kernel<1, 1, 0, 0><<<grid, dim3(32), 0, stream>>>(
                    x, pw.ap, p0, p1, out, Cin, Cout, Hin, Win, wshift, npix, stride, pad, pw.kt, pw.mt);
        } else {
            if (epi == 1)
                conv_wmma_kernel<3, 3, 0, 1><<<grid, dim3(32), 0, stream>>>(
                    x, pw.ap, p0, p1, out, Cin, Cout, Hin, Win, wshift, npix, stride, pad, pw.kt, pw.mt);
            else if (relu_in)
                conv_wmma_kernel<3, 3, 1, 0><<<grid, dim3(32), 0, stream>>>(
                    x, pw.ap, p0, p1, out, Cin, Cout, Hin, Win, wshift, npix, stride, pad, pw.kt, pw.mt);
            else
                conv_wmma_kernel<3, 3, 0, 0><<<grid, dim3(32), 0, stream>>>(
                    x, pw.ap, p0, p1, out, Cin, Cout, Hin, Win, wshift, npix, stride, pad, pw.kt, pw.mt);
        }
    };

    // ---- pack all weights (f16 A-fragments) ----
    PW pw_in3 = pack(in_w3, 64, 40, 1, 1);
    PW pw_in4 = pack(in_w4, 64, 80, 1, 1);
    PW pw_in5 = pack(in_w5, 64, 192, 1, 1);
    PW pw_in6 = pack(in_w6, 64, 192, 3, 3);
    PW pw_in7 = pack(in_w7, 64, 64, 3, 3);
    PW pw_td[2][4], pw_oc[2][5];
    for (int blk = 0; blk < 2; ++blk) {
        for (int l = 0; l < 4; ++l)
            pw_td[blk][l] = pack(td_w + (size_t)(blk * 5 + l) * 4096, 64, 64, 1, 1);
        for (int l = 1; l < 5; ++l)
            pw_oc[blk][l] = pack(oc_w + (size_t)(blk * 5 + l) * 4096, 64, 64, 1, 1);
    }
    PW pw_cls[3], pw_reg[3];
    for (int i = 0; i < 3; ++i) {
        pw_cls[i] = pack(cls_w + (size_t)i * 36864, 64, 64, 3, 3);
        pw_reg[i] = pack(reg_w + (size_t)i * 36864, 64, 64, 3, 3);
    }
    PW pw_cout = pack(cls_ow, 720, 64, 3, 3);
    PW pw_rout = pack(reg_ow, 36, 64, 3, 3);

    // ---- BN folding ----
    float* td_scale = (float*)alloc(640 * 4);
    float* td_shift = (float*)alloc(640 * 4);
    float* oc_scale = (float*)alloc(640 * 4);
    float* oc_shift = (float*)alloc(640 * 4);
    bnprep_kernel<<<3, 256, 0, stream>>>(td_b, td_g, td_beta, td_m, td_v, td_scale, td_shift, 640);
    bnprep_kernel<<<3, 256, 0, stream>>>(oc_b, oc_g, oc_beta, oc_m, oc_v, oc_scale, oc_shift, 640);

    // ---- feature buffers (image 0 only; levels 64,32,16,8,4) ----
    const int fsz[5] = {64, 32, 16, 8, 4};
    float *featA[5], *featB[5], *tdb[5] = {nullptr, nullptr, nullptr, nullptr, nullptr};
    for (int l = 0; l < 5; ++l) {
        size_t n = (size_t)64 * fsz[l] * fsz[l];
        featA[l] = (float*)alloc(n * 4);
        featB[l] = (float*)alloc(n * 4);
    }
    for (int l = 1; l < 4; ++l)
        tdb[l] = (float*)alloc((size_t)64 * fsz[l] * fsz[l] * 4);
    float* fuse_tmp = (float*)alloc((size_t)64 * 64 * 64 * 4);
    float* h0 = (float*)alloc((size_t)64 * 64 * 64 * 4);
    float* h1 = (float*)alloc((size_t)64 * 64 * 64 * 4);
    float *rout[5], *coutb[5];
    for (int l = 0; l < 5; ++l) {
        rout[l]  = (float*)alloc((size_t)36  * fsz[l] * fsz[l] * 4);
        coutb[l] = (float*)alloc((size_t)720 * fsz[l] * fsz[l] * 4);
    }

    // ---- FPN input projections (batch element 0) ----
    conv(c3, pw_in3, in_b3, nullptr, featA[0], 40, 64, 64, 64, 64, 64, 1, 1, 1, 0, 0, 0);
    conv(c4, pw_in4, in_b4, nullptr, featA[1], 80, 64, 32, 32, 32, 32, 1, 1, 1, 0, 0, 0);
    conv(c5, pw_in5, in_b5, nullptr, featA[2], 192, 64, 16, 16, 16, 16, 1, 1, 1, 0, 0, 0);
    conv(c5, pw_in6, in_b6, nullptr, featA[3], 192, 64, 16, 16, 8, 8, 3, 3, 2, 1, 0, 0);
    conv(featA[3], pw_in7, in_b7, nullptr, featA[4], 64, 64, 8, 8, 4, 4, 3, 3, 2, 1, 1, 0);

    // ---- BiFPN x2 ----
    float *cur[5], *nxt[5];
    for (int l = 0; l < 5; ++l) { cur[l] = featA[l]; nxt[l] = featB[l]; }
    for (int blk = 0; blk < 2; ++blk) {
        const float* w1p = fpn_w1 + blk * 2;
        const float* w2p = fpn_w2 + blk * 3;
        // top-down (all td fusions use original feats)
        for (int l = 3; l >= 0; --l) {
            int f = fsz[l]; int n = 64 * f * f;
            fuse2_kernel<<<dim3((n + 255) / 256), dim3(256), 0, stream>>>(
                cur[l], cur[l + 1], fuse_tmp, w1p, f, f);
            float* dst = (l == 0) ? nxt[0] : tdb[l];   // p3td is also next-feats[0]
            int idx = blk * 5 + l;
            conv(fuse_tmp, pw_td[blk][l], td_scale + idx * 64, td_shift + idx * 64, dst,
                 64, 64, f, f, f, f, 1, 1, 1, 0, 0, 2);
        }
        // bottom-up outputs
        for (int l = 1; l <= 4; ++l) {
            int f = fsz[l]; int n = 64 * f * f;
            const float* bmid = (l == 4) ? cur[4] : tdb[l];           // p7td == p7x
            const float* clow = (l == 1) ? nxt[0] : tdb[l - 1];
            fuse3_kernel<<<dim3((n + 255) / 256), dim3(256), 0, stream>>>(
                cur[l], bmid, clow, fuse_tmp, w2p, f, f);
            int idx = blk * 5 + l;
            conv(fuse_tmp, pw_oc[blk][l], oc_scale + idx * 64, oc_shift + idx * 64, nxt[l],
                 64, 64, f, f, f, f, 1, 1, 1, 0, 0, 2);
        }
        for (int l = 0; l < 5; ++l) { float* t = cur[l]; cur[l] = nxt[l]; nxt[l] = t; }
    }

    // ---- detection heads ----
    for (int l = 0; l < 5; ++l) {
        int f = fsz[l];
        // regression head
        conv(cur[l], pw_reg[0], reg_b + 0,   nullptr, h0, 64, 64, f, f, f, f, 3, 3, 1, 1, 0, 1);
        conv(h0,     pw_reg[1], reg_b + 64,  nullptr, h1, 64, 64, f, f, f, f, 3, 3, 1, 1, 0, 1);
        conv(h1,     pw_reg[2], reg_b + 128, nullptr, h0, 64, 64, f, f, f, f, 3, 3, 1, 1, 0, 1);
        conv(h0,     pw_rout,   reg_ob,      nullptr, rout[l], 64, 36, f, f, f, f, 3, 3, 1, 1, 0, 0);
        // classification head (logits only; sigmoid folded into decode)
        conv(cur[l], pw_cls[0], cls_b + 0,   nullptr, h0, 64, 64, f, f, f, f, 3, 3, 1, 1, 0, 1);
        conv(h0,     pw_cls[1], cls_b + 64,  nullptr, h1, 64, 64, f, f, f, f, 3, 3, 1, 1, 0, 1);
        conv(h1,     pw_cls[2], cls_b + 128, nullptr, h0, 64, 64, f, f, f, f, 3, 3, 1, 1, 0, 1);
        conv(h0,     pw_cout,   cls_ob,      nullptr, coutb[l], 64, 720, f, f, f, f, 3, 3, 1, 1, 0, 0);
    }

    // ---- decode / top-k / NMS ----
    float* boxes  = (float*)alloc((size_t)NA * 4 * 4);
    float* scores = (float*)alloc((size_t)NA * 4);
    int*   labels = (int*)  alloc((size_t)NA * 4);
    float* work   = (float*)alloc((size_t)NA * 4);
    float* tops   = (float*)alloc((size_t)TOPK * 4);
    int*   topi   = (int*)  alloc((size_t)TOPK * 4);

    decode_kernel<<<dim3((NA + 255) / 256), dim3(256), 0, stream>>>(
        rout[0], rout[1], rout[2], rout[3], rout[4],
        coutb[0], coutb[1], coutb[2], coutb[3], coutb[4],
        boxes, scores, labels);
    topk_kernel<<<1, 256, 0, stream>>>(scores, work, tops, topi);
    nms_kernel<<<1, 1024, 0, stream>>>(boxes, labels, tops, topi, (float*)d_out);
}
